// EdgeGCNLayer_39367670235755
// MI455X (gfx1250) — matrix operand order repaired and checked by reading, not compiled, not run on hardware
//
#include <hip/hip_runtime.h>

// ---------------- problem constants ----------------
#define B_   8
#define V_   10000
#define F_   128
#define E_   160000
#define O_   128
#define ROWS (B_ * V_)          // 80000 rows of the (BV,F) LHS
#define BN_EPS 1e-5f

typedef _Float16 v16h __attribute__((ext_vector_type(16)));
typedef _Float16 h8   __attribute__((ext_vector_type(8)));
typedef float    v8f  __attribute__((ext_vector_type(8)));
typedef float    f4   __attribute__((ext_vector_type(4)));

__device__ __forceinline__ void atomAddF(float* p, float v) {
    unsafeAtomicAdd(p, v);      // lowers to global_atomic_add_f32 (no return)
}

// ---------------- kernel 0: zero the stats accumulators ----------------
__global__ __launch_bounds__(256) void k_zero_stats(float* st) {
    st[threadIdx.x] = 0.0f;     // 256 floats: sum[128], sumsq[128]
}

// ---------------- kernel 1: convert weights f32 -> f16 ----------------
__global__ __launch_bounds__(256) void k_cvt_w(const float* __restrict__ Wself,
                                               const float* __restrict__ Wnode,
                                               _Float16* __restrict__ WhS,
                                               _Float16* __restrict__ WhN) {
    int i = blockIdx.x * 256 + threadIdx.x;
    if (i < O_ * F_) {
        WhS[i] = (_Float16)Wself[i];
        WhN[i] = (_Float16)Wnode[i];
    }
}

// ---------------- kernel 2: fused dual GEMM via WMMA ----------------
// One block = 128 threads = 4 waves. Block computes a 16-row strip of the
// (ROWS x 128) outputs; wave w owns columns [32w, 32w+32) (two 16x16 n-tiles)
// for BOTH H (=X W_self^T + b) and Xn (=X W_node^T). K loop: 4 x K=32.
__global__ __launch_bounds__(128) void k_gemm_dual(const float* __restrict__ X,
                                                   const _Float16* __restrict__ WhS,
                                                   const _Float16* __restrict__ WhN,
                                                   const float* __restrict__ bself,
                                                   float* __restrict__ H,
                                                   float* __restrict__ Xn) {
    const int row0 = blockIdx.x * 16;
    const int wave = threadIdx.x >> 5;
    const int lane = threadIdx.x & 31;
    const int r    = lane & 15;      // row (A) / column (B) within tile
    const int hi   = lane >> 4;      // half-wave selector
    const int n0b  = wave * 32;

    v8f accS0 = {}, accS1 = {}, accN0 = {}, accN1 = {};

#pragma unroll
    for (int kt = 0; kt < 4; ++kt) {
        const int k0 = kt * 32;

        // ---- A fragment: 16x32 f16, ISA layout:
        // lane<16: K = k0+[0..7] and k0+16+[0..7]; lane>=16: +8 on both chunks.
        v16h a;
        {
            const float* xp = X + (size_t)(row0 + r) * F_ + k0 + hi * 8;
            f4 x0 = *(const f4*)(xp);
            f4 x1 = *(const f4*)(xp + 4);
            f4 x2 = *(const f4*)(xp + 16);
            f4 x3 = *(const f4*)(xp + 20);
            float t[16];
            *(f4*)(t + 0)  = x0; *(f4*)(t + 4)  = x1;
            *(f4*)(t + 8)  = x2; *(f4*)(t + 12) = x3;
#pragma unroll
            for (int i = 0; i < 16; ++i) a[i] = (_Float16)t[i];
        }

        // ---- B fragments: 32x16 f16; lane = column n, same K order as A:
        // lane<16: K = k0+[0..7] and k0+16+[0..7]; lane>=16: +8 on both chunks.
        v16h bS[2], bN[2];
#pragma unroll
        for (int j = 0; j < 2; ++j) {
            const int n0 = n0b + j * 16;
            const _Float16* ws = WhS + (size_t)(n0 + r) * F_ + k0 + hi * 8;
            const _Float16* wn = WhN + (size_t)(n0 + r) * F_ + k0 + hi * 8;
            h8 s0 = *(const h8*)(ws);
            h8 s1 = *(const h8*)(ws + 16);
            h8 n0v = *(const h8*)(wn);
            h8 n1v = *(const h8*)(wn + 16);
#pragma unroll
            for (int i = 0; i < 8; ++i) {
                bS[j][i] = s0[i];  bS[j][8 + i] = s1[i];
                bN[j][i] = n0v[i]; bN[j][8 + i] = n1v[i];
            }
        }

        accS0 = __builtin_amdgcn_wmma_f32_16x16x32_f16(false, a, false, bS[0],
                                                       (short)0, accS0, false, false);
        accS1 = __builtin_amdgcn_wmma_f32_16x16x32_f16(false, a, false, bS[1],
                                                       (short)0, accS1, false, false);
        accN0 = __builtin_amdgcn_wmma_f32_16x16x32_f16(false, a, false, bN[0],
                                                       (short)0, accN0, false, false);
        accN1 = __builtin_amdgcn_wmma_f32_16x16x32_f16(false, a, false, bN[1],
                                                       (short)0, accN1, false, false);
    }

    // ---- epilogue: C/D layout: lane<16 -> M=v, N=lane; lane>=16 -> M=8+v.
    const int n  = lane & 15;
    const int mb = hi * 8;
#pragma unroll
    for (int j = 0; j < 2; ++j) {
        const int nc = n0b + j * 16 + n;
        const float bs = bself[nc];
        const v8f aS = j ? accS1 : accS0;
        const v8f aN = j ? accN1 : accN0;
#pragma unroll
        for (int v = 0; v < 8; ++v) {
            const size_t row = (size_t)(row0 + mb + v);
            H[row * O_ + nc]  = aS[v] + bs;
            Xn[row * O_ + nc] = aN[v];
        }
    }
}

// ---------------- kernel 3: edge gather + scatter-add ----------------
// One wave per edge; 32 lanes x float4 cover the 128 channels; loop batches.
__global__ __launch_bounds__(256) void k_edge(const float* __restrict__ Xn,
                                              const int* __restrict__ ei,
                                              const float* __restrict__ ea,
                                              const float* __restrict__ Wedge,
                                              float* __restrict__ H) {
    const int e = blockIdx.x * 8 + (threadIdx.x >> 5);
    if (e >= E_) return;
    const int lane = threadIdx.x & 31;
    const int src = ei[2 * e + 0];
    const int dst = ei[2 * e + 1];
    const f4 we = *(const f4*)(Wedge + lane * 4);

#pragma unroll
    for (int b = 0; b < B_; ++b) {
        const float a = ea[(size_t)b * E_ + e];
        const f4 x = *(const f4*)(Xn + ((size_t)b * V_ + src) * O_ + lane * 4);
        float* hp = H + ((size_t)b * V_ + dst) * O_ + lane * 4;
        atomAddF(hp + 0, x[0] + a * we[0]);
        atomAddF(hp + 1, x[1] + a * we[1]);
        atomAddF(hp + 2, x[2] + a * we[2]);
        atomAddF(hp + 3, x[3] + a * we[3]);
    }
}

// ---------------- kernel 4: per-channel sum / sumsq reduction ----------------
__global__ __launch_bounds__(128) void k_stats(const float* __restrict__ H,
                                               float* __restrict__ st) {
    const int c = threadIdx.x;                 // channel 0..127
    float s = 0.f, s2 = 0.f;
    for (int row = blockIdx.x; row < ROWS; row += gridDim.x) {
        const float v = H[(size_t)row * O_ + c];
        s += v; s2 += v * v;
    }
    atomAddF(&st[c], s);
    atomAddF(&st[O_ + c], s2);
}

// ---------------- kernel 5: fold BN stats into scale/shift ----------------
__global__ __launch_bounds__(128) void k_finalize(const float* __restrict__ st,
                                                  const float* __restrict__ gamma,
                                                  const float* __restrict__ beta,
                                                  float* __restrict__ sc,
                                                  float* __restrict__ sh) {
    const int c = threadIdx.x;
    const float invN = 1.0f / (float)ROWS;
    const float mean = st[c] * invN;
    const float var  = st[O_ + c] * invN - mean * mean;
    const float s    = gamma[c] * __frsqrt_rn(var + BN_EPS);
    sc[c] = s;
    sh[c] = beta[c] - mean * s;
}

// ---------------- kernel 6: BN apply + ReLU, in place on d_out ----------------
__global__ __launch_bounds__(256) void k_bnrelu(float* __restrict__ H,
                                                const float* __restrict__ sc,
                                                const float* __restrict__ sh) {
    const size_t idx = (size_t)blockIdx.x * 256 + threadIdx.x;   // float4 index
    const int c4 = ((int)(idx & 31)) * 4;                        // 128/4 = 32
    f4 h = ((const f4*)H)[idx];
    f4 o;
#pragma unroll
    for (int i = 0; i < 4; ++i) {
        const float v = h[i] * sc[c4 + i] + sh[c4 + i];
        o[i] = v > 0.f ? v : 0.f;
    }
    ((f4*)H)[idx] = o;
}

// ---------------- launcher ----------------
extern "C" void kernel_launch(void* const* d_in, const int* in_sizes, int n_in,
                              void* d_out, int out_size, void* d_ws, size_t ws_size,
                              hipStream_t stream) {
    const float* X     = (const float*)d_in[0];
    const int*   ei    = (const int*)d_in[1];
    const float* ea    = (const float*)d_in[2];
    const float* Wnode = (const float*)d_in[3];
    const float* Wedge = (const float*)d_in[4];
    const float* Wself = (const float*)d_in[5];
    const float* bself = (const float*)d_in[6];
    const float* gamma = (const float*)d_in[7];
    const float* beta  = (const float*)d_in[8];
    float* H = (float*)d_out;                     // (B,V,O) f32

    // workspace layout
    char* ws = (char*)d_ws;
    const size_t XN_BYTES = (size_t)ROWS * O_ * sizeof(float);   // 40.96 MB
    float*    Xn  = (float*)ws;
    _Float16* WhS = (_Float16*)(ws + XN_BYTES);
    _Float16* WhN = WhS + O_ * F_;
    float*    st  = (float*)(ws + XN_BYTES + 2 * O_ * F_ * sizeof(_Float16));
    float*    sc  = st + 2 * O_;
    float*    sh  = sc + O_;

    k_zero_stats<<<1, 256, 0, stream>>>(st);
    k_cvt_w<<<(O_ * F_ + 255) / 256, 256, 0, stream>>>(Wself, Wnode, WhS, WhN);
    k_gemm_dual<<<ROWS / 16, 128, 0, stream>>>(X, WhS, WhN, bself, H, Xn);
    k_edge<<<E_ / 8, 256, 0, stream>>>(Xn, ei, ea, Wedge, H);
    k_stats<<<256, 128, 0, stream>>>(H, st);
    k_finalize<<<1, 128, 0, stream>>>(st, gamma, beta, sc, sh);
    k_bnrelu<<<(ROWS * O_ / 4) / 256, 256, 0, stream>>>(H, sc, sh);
}
